// Net_32555852104135
// MI455X (gfx1250) — compile-verified, matrix operand order checked by reading, not run on hardware
//
#include <hip/hip_runtime.h>
#include <hip/hip_bf16.h>
#include <stdint.h>

// ---------------- problem constants ----------------
#define NNODES 20000
#define NEDGES 320000
#define NGRAPH 512
#define FEAT   202
#define C1D    64
#define FLATD  (FEAT*C1D)          // 12928
// padded K dims (multiples of 32)
#define KP_NM  224                 // enet input (202 -> 224)
#define KP_NB  416                 // edgenet / snet input (404/405 -> 416)

typedef __bf16   bf16x16 __attribute__((ext_vector_type(16)));
typedef float    f32x8   __attribute__((ext_vector_type(8)));
typedef uint32_t u32x4   __attribute__((ext_vector_type(4)));

__device__ __forceinline__ uint16_t f2bf(float x) {
  union { float f; uint32_t u; } a; a.f = x;
  uint32_t r = a.u + 0x7FFFu + ((a.u >> 16) & 1u);
  return (uint16_t)(r >> 16);
}
__device__ __forceinline__ float bf2f(uint16_t h) {
  union { uint32_t u; float f; } a; a.u = ((uint32_t)h) << 16; return a.f;
}

// ---------------- elementwise / scatter kernels ----------------
__global__ void k_zero(float* p, int n) {
  int i = blockIdx.x * blockDim.x + threadIdx.x;
  if (i < n) p[i] = 0.f;
}

__global__ void k_deg(const int* __restrict__ ii, float* deg, int E) {
  int e = blockIdx.x * blockDim.x + threadIdx.x;
  if (e < E) atomicAdd(&deg[ii[e]], 1.f);
}

// nm = 0.5*(x_i + x_j); write bf16 into Anm[:, :202] (pad 202..223 = 0)
// and Anb[:, :202]; atomic-accumulate nm into ga (scatter-mean numerator)
__global__ void k_nm(const float* __restrict__ x, const int* __restrict__ ii,
                     const int* __restrict__ jj, float* __restrict__ ga,
                     uint16_t* __restrict__ Anm, uint16_t* __restrict__ Anb, int E) {
  int id = blockIdx.x * blockDim.x + threadIdx.x;
  if (id >= E * KP_NM) return;
  int f = id % KP_NM, e = id / KP_NM;
  if (f < FEAT) {
    int i = ii[e], j = jj[e];
    float nm = 0.5f * (x[(size_t)i * FEAT + f] + x[(size_t)j * FEAT + f]);
    uint16_t b = f2bf(nm);
    Anm[(size_t)e * KP_NM + f] = b;
    Anb[(size_t)e * KP_NB + f] = b;
    atomicAdd(&ga[(size_t)i * FEAT + f], nm);
  } else {
    Anm[(size_t)e * KP_NM + f] = 0;
  }
}

__global__ void k_gadiv(float* __restrict__ ga, const float* __restrict__ deg, int n) {
  int id = blockIdx.x * blockDim.x + threadIdx.x;
  if (id >= n) return;
  int node = id / FEAT;
  ga[id] = ga[id] / fmaxf(deg[node], 1.f);
}

// Anb[:, 202:404] = bf16(ga[index_i]); Anb[:, 404:416] = 0
__global__ void k_nbgather(const float* __restrict__ ga, const int* __restrict__ ii,
                           uint16_t* __restrict__ Anb, int E) {
  int id = blockIdx.x * blockDim.x + threadIdx.x;
  if (id >= E * 214) return;
  int t = id % 214, e = id / 214;
  int c = 202 + t;
  uint16_t v = 0;
  if (t < FEAT) v = f2bf(ga[(size_t)ii[e] * FEAT + t]);
  Anb[(size_t)e * KP_NB + c] = v;
}

// z = [x_i | x_j | edge_scalar | pad]
__global__ void k_zbuild(const float* __restrict__ x, const int* __restrict__ ii,
                         const int* __restrict__ jj, const float* __restrict__ edge1,
                         uint16_t* __restrict__ Az, int E) {
  int id = blockIdx.x * blockDim.x + threadIdx.x;
  if (id >= E * KP_NB) return;
  int c = id % KP_NB, e = id / KP_NB;
  float v;
  if (c < FEAT)            v = x[(size_t)ii[e] * FEAT + c];
  else if (c < 2 * FEAT)   v = x[(size_t)jj[e] * FEAT + (c - FEAT)];
  else if (c == 2 * FEAT)  v = edge1[e];
  else                     v = 0.f;
  Az[(size_t)e * KP_NB + c] = f2bf(v);
}

__global__ void k_sm_ex(float* __restrict__ buf, float* __restrict__ den,
                        const int* __restrict__ ii, int E) {
  int e = blockIdx.x * blockDim.x + threadIdx.x;
  if (e >= E) return;
  float ex = expf(buf[e]);
  buf[e] = ex;
  atomicAdd(&den[ii[e]], ex);
}
__global__ void k_sm_div(float* __restrict__ buf, const float* __restrict__ den,
                         const int* __restrict__ ii, int E) {
  int e = blockIdx.x * blockDim.x + threadIdx.x;
  if (e >= E) return;
  buf[e] = buf[e] / (den[ii[e]] + 1e-9f);
}

__global__ void k_msgscatter(const float* __restrict__ msg, const float* __restrict__ aij,
                             const int* __restrict__ ii, float* __restrict__ emb, int E) {
  int id = blockIdx.x * blockDim.x + threadIdx.x;
  if (id >= E * FEAT) return;
  int f = id % FEAT, e = id / FEAT;
  atomicAdd(&emb[(size_t)ii[e] * FEAT + f], aij[e] * msg[(size_t)e * FEAT + f]);
}

__global__ void k_xupdate(const float* __restrict__ xin, const float* __restrict__ emb,
                          const float* __restrict__ deg, float* __restrict__ xout, int n) {
  int id = blockIdx.x * blockDim.x + threadIdx.x;
  if (id >= n) return;
  int node = id / FEAT;
  xout[id] = xin[id] + emb[id] / fmaxf(deg[node], 1.f);
}

// ---------------- attention pooling ----------------
__global__ void k_pool_logit(const float* __restrict__ x, const float* __restrict__ w,
                             const int* __restrict__ seg, float* __restrict__ exbuf,
                             float* __restrict__ den, int n) {
  int node = blockIdx.x * blockDim.x + threadIdx.x;
  if (node >= n) return;
  float s = 0.f;
  for (int f = 0; f < FEAT; ++f) s += x[(size_t)node * FEAT + f] * w[f];
  float ex = expf(s);
  exbuf[node] = ex;
  atomicAdd(&den[seg[node]], ex);
}
__global__ void k_pool_accum(const float* __restrict__ x, const float* __restrict__ exbuf,
                             const int* __restrict__ seg, float* __restrict__ gp, int n) {
  int id = blockIdx.x * blockDim.x + threadIdx.x;
  if (id >= n * FEAT) return;
  int f = id % FEAT, node = id / FEAT;
  atomicAdd(&gp[(size_t)seg[node] * FEAT + f], exbuf[node] * x[id]);
}
__global__ void k_pool_norm(float* __restrict__ gp, const float* __restrict__ den, int n) {
  int id = blockIdx.x * blockDim.x + threadIdx.x;
  if (id >= n) return;
  gp[id] = gp[id] / (den[id / FEAT] + 1e-9f);
}

// ---------------- conv1d (SAME, k=3) -> bf16 into x3b and A4[:,1024:] ----------------
__global__ void k_conv1d(const float* __restrict__ xgp, const float* __restrict__ w,
                         const float* __restrict__ b, uint16_t* __restrict__ x3b,
                         uint16_t* __restrict__ A4) {
  int id = blockIdx.x * blockDim.x + threadIdx.x;
  if (id >= NGRAPH * FEAT * C1D) return;
  int c = id % C1D, p = (id / C1D) % FEAT, g = id / (C1D * FEAT);
  float acc = b[c];
#pragma unroll
  for (int t = 0; t < 3; ++t) {
    int q = p + t - 1;
    if (q >= 0 && q < FEAT) acc += xgp[(size_t)g * FEAT + q] * w[t * C1D + c];
  }
  acc = fmaxf(acc, 0.f);
  uint16_t v = f2bf(acc);
  x3b[(size_t)g * FLATD + p * C1D + c] = v;
  A4[(size_t)g * 13952 + 1024 + p * C1D + c] = v;
}

// concat-fill: dst[r, coloff+c] = bf16(src[r, c]) for c<srccols else 0  (c<totcols)
__global__ void k_fillb(const float* __restrict__ src, int srccols,
                        uint16_t* __restrict__ dst, int ldd, int coloff,
                        int totcols, int rows) {
  int id = blockIdx.x * blockDim.x + threadIdx.x;
  if (id >= rows * totcols) return;
  int c = id % totcols, r = id / totcols;
  float v = (c < srccols) ? src[(size_t)r * srccols + c] : 0.f;
  dst[(size_t)r * ldd + coloff + c] = f2bf(v);
}

// ---------------- WMMA weight pre-pack ----------------
// B-fragment layout (16x16x32 bf16, wave32): lane l holds column n = nt*16+(l&15);
// h=l>>4; dword v<4 -> k pair (kt*32 + h*8 + 2v), v>=4 -> (kt*32 + 16 + h*8 + 2(v-4)).
__global__ void k_pack(const float* __restrict__ W, uint32_t* __restrict__ Wp,
                       int K, int N, int Ntp, int Kt) {
  int id = blockIdx.x * blockDim.x + threadIdx.x;
  if (id >= Kt * Ntp * 256) return;
  int v = id & 7;
  int lane = (id >> 3) & 31;
  int nt = (id >> 8) % Ntp;
  int kt = (id >> 8) / Ntp;
  int n = nt * 16 + (lane & 15);
  int h = lane >> 4;
  int k = kt * 32 + ((v < 4) ? (h * 8 + 2 * v) : (16 + h * 8 + 2 * (v - 4)));
  float w0 = (k     < K && n < N) ? W[(size_t)k * N + n]       : 0.f;
  float w1 = (k + 1 < K && n < N) ? W[(size_t)(k + 1) * N + n] : 0.f;
  Wp[id] = (uint32_t)f2bf(w0) | ((uint32_t)f2bf(w1) << 16);
}

// ---------------- WMMA GEMM: C[M,N] = act(A[M,Kp]_bf16 @ Wpack + bias) ----------------
// block = 128 threads (4 waves); each wave computes a 16x128 strip (8 n-tiles).
// B tile (32x128 bf16, 8KB, fragment-ordered) is double-buffered in LDS via
// global_load_async_to_lds_b128 (ASYNCcnt); the copy of k-tile kt+1 overlaps the
// 8 WMMAs of tile kt and all 4 waves share one copy of B.  The k-loop is manually
// unrolled by 2 so LDS buffer indices and the A-fragment double-buffer registers
// (af0/af1) are compile-time constants (no dynamic indexing -> stays in VGPRs).
// NTA: non-temporal A loads (streaming A must not evict L2-resident x/weights).
template<bool NTA>
__global__ void k_gemm_t(const uint16_t* __restrict__ A, int lda,
                         const uint32_t* __restrict__ Wp, const float* __restrict__ bias,
                         int M, int N, int Ntp, int Kt,
                         float* Cf, int ldcf, uint16_t* Cb, int ldcb, int relu) {
  __shared__ u32x4 Bsm[2][512];                 // 2 x 8KB
  int tid  = threadIdx.x;
  int lane = tid & 31;
  int wid  = tid >> 5;
  int mbase = (blockIdx.x * 4 + wid) << 4;
  bool active = (mbase < M);                    // wave-uniform
  int nt0 = blockIdx.y << 3;
  int h = lane >> 4;
  int mrow = mbase + (lane & 15);
  if (mrow >= M) mrow = M - 1;
  const u32x4* Arow = (const u32x4*)(A + (size_t)mrow * (size_t)lda);
  const u32x4* Wp4  = (const u32x4*)Wp;

  union frag { bf16x16 v; u32x4 q[2]; };

  f32x8 acc[8];
#pragma unroll
  for (int t = 0; t < 8; ++t) { f32x8 z = {0.f,0.f,0.f,0.f,0.f,0.f,0.f,0.f}; acc[t] = z; }

  // async-copy one 8KB B tile (512 x 16B) into Bsm[buf]; 128 threads x 4 chunks
  auto stage = [&](int kt, int buf) {
    const u32x4* src = Wp4 + (size_t)(kt * Ntp + nt0) * 64 + tid;
#pragma unroll
    for (int r = 0; r < 4; ++r) {
      uint32_t dst = (uint32_t)(uintptr_t)&Bsm[buf][tid + r * 128]; // LDS offset = addr[31:0]
      asm volatile("global_load_async_to_lds_b128 %0, %1, off"
                   :: "v"(dst), "v"(src + r * 128) : "memory");
    }
  };
  auto loadA = [&](int kt, frag& f) {
    if (NTA) {
      f.q[0] = __builtin_nontemporal_load(&Arow[kt * 4 + h]);
      f.q[1] = __builtin_nontemporal_load(&Arow[kt * 4 + 2 + h]);
    } else {
      f.q[0] = Arow[kt * 4 + h];
      f.q[1] = Arow[kt * 4 + 2 + h];
    }
  };
  // 8 WMMAs on one k-tile; buf is a literal at every call site -> static LDS addrs
  auto compute = [&](const frag& a, int buf) {
    frag b0, b1;
    b0.q[0] = Bsm[buf][lane * 2];
    b0.q[1] = Bsm[buf][lane * 2 + 1];
#pragma unroll
    for (int t = 0; t < 8; ++t) {
      if (t < 7) {                              // ds_loads for tile t+1 in flight
        frag& nb = (t & 1) ? b0 : b1;
        nb.q[0] = Bsm[buf][((t + 1) * 32 + lane) * 2];
        nb.q[1] = Bsm[buf][((t + 1) * 32 + lane) * 2 + 1];
      }
      const frag& cb = (t & 1) ? b1 : b0;
      acc[t] = __builtin_amdgcn_wmma_f32_16x16x32_bf16(
          false, a.v, false, cb.v, (short)0, acc[t], false, false);
    }
  };

  frag af0, af1;
  if (active) loadA(0, af0);
  stage(0, 0);
  asm volatile("s_wait_asynccnt 0x0" ::: "memory");
  __syncthreads();

  int kt = 0;
  while (kt < Kt) {
    // even step: compute Bsm[0]/af0, prefetch kt+1 -> Bsm[1]/af1
    if (kt + 1 < Kt) {
      stage(kt + 1, 1);
      if (active) loadA(kt + 1, af1);
    }
    if (active) compute(af0, 0);
    if (kt + 1 < Kt) {
      asm volatile("s_wait_asynccnt 0x0" ::: "memory");
      __syncthreads();
    }
    ++kt;
    if (kt >= Kt) break;
    // odd step: compute Bsm[1]/af1, prefetch kt+1 -> Bsm[0]/af0
    if (kt + 1 < Kt) {
      stage(kt + 1, 0);
      if (active) loadA(kt + 1, af0);
    }
    if (active) compute(af1, 1);
    if (kt + 1 < Kt) {
      asm volatile("s_wait_asynccnt 0x0" ::: "memory");
      __syncthreads();
    }
    ++kt;
  }

  if (!active) return;
#pragma unroll
  for (int t = 0; t < 8; ++t) {
    int col = ((nt0 + t) << 4) + (lane & 15);
    if (col >= N) continue;
    float bv = bias ? bias[col] : 0.f;
#pragma unroll
    for (int r = 0; r < 8; ++r) {               // C frag: VGPR r -> row mbase + 8h + r
      int row = mbase + h * 8 + r;
      if (row >= M) continue;
      float v = acc[t][r] + bv;
      if (relu) v = fmaxf(v, 0.f);
      if (Cf) Cf[(size_t)row * ldcf + col] = v;
      if (Cb) Cb[(size_t)row * ldcb + col] = f2bf(v);
    }
  }
}

// ---------------- small-N dense layer (dot) ----------------
__global__ void k_dot(const uint16_t* __restrict__ A, int lda,
                      const float* __restrict__ W, const float* __restrict__ bias,
                      int M, int N, int K, float* __restrict__ out, int ldo, int relu) {
  int id = blockIdx.x * blockDim.x + threadIdx.x;
  if (id >= M * N) return;
  int n = id % N, m = id / N;
  float s = bias[n];
  for (int k = 0; k < K; ++k) s += bf2f(A[(size_t)m * lda + k]) * W[(size_t)k * N + n];
  if (relu) s = fmaxf(s, 0.f);
  out[(size_t)m * ldo + n] = s;
}

// ---------------- host side ----------------
#define LAUNCH1(kern, n, ...) do { int _n = (int)(n); \
  kern<<<(_n + 255) / 256, 256, 0, stream>>>(__VA_ARGS__); } while (0)

static void do_gemm(hipStream_t stream, const uint16_t* A, int lda, uint32_t* wpack,
                    const float* bias, int M, int N, int Ntp, int Kt,
                    float* Cf, int ldcf, uint16_t* Cb, int ldcb, int relu, bool nta) {
  dim3 g((M + 63) / 64, Ntp / 8);
  if (nta)
    k_gemm_t<true ><<<g, 128, 0, stream>>>(A, lda, wpack, bias, M, N, Ntp, Kt, Cf, ldcf, Cb, ldcb, relu);
  else
    k_gemm_t<false><<<g, 128, 0, stream>>>(A, lda, wpack, bias, M, N, Ntp, Kt, Cf, ldcf, Cb, ldcb, relu);
}
static void do_pack(hipStream_t stream, const float* W, uint32_t* wpack,
                    int K, int N, int Ntp, int Kt) {
  int total = Kt * Ntp * 256;
  k_pack<<<(total + 255) / 256, 256, 0, stream>>>(W, wpack, K, N, Ntp, Kt);
}

// one crystal_conv layer; Lc = 18 leaves [eg_w1,eg_b1,eg_w2,eg_b2,eg_w3,eg_b3, en_*, sn_*]
static void run_conv(hipStream_t stream, const float* xin, float* xout,
                     const float* const* Lc, const int* ii, const int* jj,
                     float* ga, float* deg, float* den,
                     uint16_t* Anm, uint16_t* Anb, uint16_t* H1b, uint16_t* H2b,
                     float* edge1, float* aij, float* msg, uint32_t* wpack) {
  const int E = NEDGES, NN = NNODES;
  LAUNCH1(k_zero, NN * FEAT, ga, NN * FEAT);
  LAUNCH1(k_zero, NN, deg, NN);
  LAUNCH1(k_zero, NN, den, NN);
  LAUNCH1(k_deg, E, ii, deg, E);
  LAUNCH1(k_nm, E * KP_NM, xin, ii, jj, ga, Anm, Anb, E);
  LAUNCH1(k_gadiv, NN * FEAT, ga, deg, NN * FEAT);
  LAUNCH1(k_nbgather, E * 214, ga, ii, Anb, E);
  // edgenet: 416 -> 128 -> 64 -> 1
  do_pack(stream, Lc[0], wpack, 2 * FEAT, 128, 8, 13);
  do_gemm(stream, Anb, KP_NB, wpack, Lc[1], E, 128, 8, 13, nullptr, 0, H1b, 128, 1, true);
  do_pack(stream, Lc[2], wpack, 128, 64, 8, 4);
  do_gemm(stream, H1b, 128, wpack, Lc[3], E, 64, 8, 4, nullptr, 0, H2b, 64, 1, true);
  LAUNCH1(k_dot, E, H2b, 64, Lc[4], Lc[5], E, 1, 64, edge1, 1, 0);
  // enet: 224 -> 128 -> 64 -> 1 ; then segment softmax -> aij
  do_pack(stream, Lc[6], wpack, FEAT, 128, 8, 7);
  do_gemm(stream, Anm, KP_NM, wpack, Lc[7], E, 128, 8, 7, nullptr, 0, H1b, 128, 1, true);
  do_pack(stream, Lc[8], wpack, 128, 64, 8, 4);
  do_gemm(stream, H1b, 128, wpack, Lc[9], E, 64, 8, 4, nullptr, 0, H2b, 64, 1, true);
  LAUNCH1(k_dot, E, H2b, 64, Lc[10], Lc[11], E, 1, 64, aij, 1, 0);
  LAUNCH1(k_sm_ex, E, aij, den, ii, E);
  LAUNCH1(k_sm_div, E, aij, den, ii, E);
  // snet: z = [x_i|x_j|edge] 416 -> 128 -> 64 -> 202
  LAUNCH1(k_zbuild, E * KP_NB, xin, ii, jj, edge1, Anb, E);   // Anb reused as Az
  do_pack(stream, Lc[12], wpack, 2 * FEAT + 1, 128, 8, 13);
  do_gemm(stream, Anb, KP_NB, wpack, Lc[13], E, 128, 8, 13, nullptr, 0, H1b, 128, 1, true);
  do_pack(stream, Lc[14], wpack, 128, 64, 8, 4);
  do_gemm(stream, H1b, 128, wpack, Lc[15], E, 64, 8, 4, nullptr, 0, H2b, 64, 1, true);
  do_pack(stream, Lc[16], wpack, 64, FEAT, 16, 2);
  do_gemm(stream, H2b, 64, wpack, Lc[17], E, FEAT, 16, 2, msg, FEAT, nullptr, 0, 0, true);
  // scatter-mean of aij*msg, residual add
  LAUNCH1(k_zero, NN * FEAT, ga, NN * FEAT);                  // ga reused as emb accum
  LAUNCH1(k_msgscatter, E * FEAT, msg, aij, ii, ga, E);
  LAUNCH1(k_xupdate, NN * FEAT, xin, ga, deg, xout, NN * FEAT);
}

static void run_pool(hipStream_t stream, const float* x, const float* w,
                     const int* seg, float* gp, float* den, float* exbuf) {
  LAUNCH1(k_zero, NGRAPH, den, NGRAPH);
  LAUNCH1(k_zero, NGRAPH * FEAT, gp, NGRAPH * FEAT);
  LAUNCH1(k_pool_logit, NNODES, x, w, seg, exbuf, den, NNODES);
  LAUNCH1(k_pool_accum, NNODES * FEAT, x, exbuf, seg, gp, NNODES);
  LAUNCH1(k_pool_norm, NGRAPH * FEAT, gp, den, NGRAPH * FEAT);
}

extern "C" void kernel_launch(void* const* d_in, const int* in_sizes, int n_in,
                              void* d_out, int out_size, void* d_ws, size_t ws_size,
                              hipStream_t stream) {
  const float* x  = (const float*)d_in[0];
  const int* ii   = (const int*)d_in[2];
  const int* jj   = (const int*)d_in[3];
  const int* bat  = (const int*)d_in[4];

  // --- param leaf mapping: handle dict-insertion vs jax sorted-key flattening ---
  // canonical: [0..17]=conv1, [18..35]=conv2, 36=pool1, 37=pool2,
  //            38=conv1d_w, 39=conv1d_b, [40..49]=d1w,d1b,...,d5w,d5b
  const float* L[50];
  bool sorted_keys = (n_in > 23) && (in_sizes[23] == 64);   // conv1d_b lands at 23 if sorted
  if (sorted_keys) {
    for (int i = 0; i < 18; ++i) L[i]      = (const float*)d_in[5 + i];
    L[39] = (const float*)d_in[23];
    L[38] = (const float*)d_in[24];
    for (int i = 0; i < 18; ++i) L[18 + i] = (const float*)d_in[25 + i];
    for (int i = 0; i < 10; ++i) L[40 + i] = (const float*)d_in[43 + i];
    L[36] = (const float*)d_in[53];
    L[37] = (const float*)d_in[54];
  } else {
    for (int i = 0; i < 36; ++i) L[i]      = (const float*)d_in[5 + i];
    L[36] = (const float*)d_in[41];
    L[37] = (const float*)d_in[42];
    L[38] = (const float*)d_in[43];
    L[39] = (const float*)d_in[44];
    for (int i = 0; i < 10; ++i) L[40 + i] = (const float*)d_in[45 + i];
  }

  // --- workspace carve (all 256B aligned) ---
  char* wp = (char*)d_ws;
  auto carve = [&](size_t bytes) -> void* {
    void* p = (void*)wp; wp += (bytes + 255) & ~(size_t)255; return p;
  };
  float*    ga    = (float*)carve((size_t)NNODES * FEAT * 4);
  float*    deg   = (float*)carve((size_t)NNODES * 4);
  float*    den   = (float*)carve((size_t)NNODES * 4);
  float*    x1    = (float*)carve((size_t)NNODES * FEAT * 4);
  float*    x2    = (float*)carve((size_t)NNODES * FEAT * 4);
  uint16_t* Anm   = (uint16_t*)carve((size_t)NEDGES * KP_NM * 2);
  uint16_t* Anb   = (uint16_t*)carve((size_t)NEDGES * KP_NB * 2);
  uint16_t* H1b   = (uint16_t*)carve((size_t)NEDGES * 128 * 2);
  uint16_t* H2b   = (uint16_t*)carve((size_t)NEDGES * 64 * 2);
  float*    edge1 = (float*)carve((size_t)NEDGES * 4);
  float*    aij   = (float*)carve((size_t)NEDGES * 4);
  float*    msg   = (float*)carve((size_t)NEDGES * FEAT * 4);
  uint32_t* wpack = (uint32_t*)carve((size_t)14 * 1024 * 1024);
  float*    x1gp  = (float*)carve((size_t)NGRAPH * FEAT * 4);
  float*    x2gp  = (float*)carve((size_t)NGRAPH * FEAT * 4);
  uint16_t* x3b   = (uint16_t*)carve((size_t)NGRAPH * FLATD * 2);
  uint16_t* A2    = (uint16_t*)carve((size_t)NGRAPH * 736 * 2);
  uint16_t* A3    = (uint16_t*)carve((size_t)NGRAPH * 1248 * 2);
  uint16_t* A4    = (uint16_t*)carve((size_t)NGRAPH * 13952 * 2);
  uint16_t* A5    = (uint16_t*)carve((size_t)NGRAPH * 256 * 2);
  (void)ws_size; (void)out_size;

  // --- two crystal_conv layers ---
  run_conv(stream, x,  x1, &L[0],  ii, jj, ga, deg, den, Anm, Anb, H1b, H2b, edge1, aij, msg, wpack);
  run_conv(stream, x1, x2, &L[18], ii, jj, ga, deg, den, Anm, Anb, H1b, H2b, edge1, aij, msg, wpack);

  // --- attention pooling (reuse aij as node-sized exp buffer) ---
  run_pool(stream, x1, L[36], bat, x1gp, den, aij);
  run_pool(stream, x2, L[37], bat, x2gp, den, aij);

  // --- conv1d + dense head ---
  LAUNCH1(k_conv1d, NGRAPH * FEAT * C1D, x2gp, L[38], L[39], x3b, A4);
  // d1: 12928 -> 512 (relu) into A2[:, 0:512]
  do_pack(stream, L[40], wpack, FLATD, 512, 32, 404);
  do_gemm(stream, x3b, FLATD, wpack, L[41], NGRAPH, 512, 32, 404, nullptr, 0, A2, 736, 1, false);
  LAUNCH1(k_fillb, NGRAPH * 224, x1gp, FEAT, A2, 736, 512, 224, NGRAPH);
  // d2: 736(=714 pad) -> 1024 (relu) into A3[:, 0:1024]
  do_pack(stream, L[42], wpack, 714, 1024, 64, 23);
  do_gemm(stream, A2, 736, wpack, L[43], NGRAPH, 1024, 64, 23, nullptr, 0, A3, 1248, 1, false);
  LAUNCH1(k_fillb, NGRAPH * 224, x2gp, FEAT, A3, 1248, 1024, 224, NGRAPH);
  // d3: 1248(=1226 pad) -> 1024 (relu) into A4[:, 0:1024]
  do_pack(stream, L[44], wpack, 1226, 1024, 64, 39);
  do_gemm(stream, A3, 1248, wpack, L[45], NGRAPH, 1024, 64, 39, nullptr, 0, A4, 13952, 1, false);
  // d4: 13952 -> 256 (relu) into A5
  do_pack(stream, L[46], wpack, 13952, 256, 16, 436);
  do_gemm(stream, A4, 13952, wpack, L[47], NGRAPH, 256, 16, 436, nullptr, 0, A5, 256, 1, false);
  // d5: 256 -> 2 (linear) -> d_out
  LAUNCH1(k_dot, NGRAPH * 2, A5, 256, L[48], L[49], NGRAPH, 2, 256, (float*)d_out, 2, 0);
}
